// SimpleGNN_68229850464790
// MI455X (gfx1250) — matrix operand order repaired and checked by reading, not compiled, not run on hardware
//
#include <hip/hip_runtime.h>
#include <hip/hip_bf16.h>

// ---------------------------------------------------------------------------
// Types for CDNA5 WMMA (wave32)
// ---------------------------------------------------------------------------
typedef __attribute__((ext_vector_type(8)))  float  v8f;
typedef __attribute__((ext_vector_type(8)))  __bf16 v8bf;
typedef __attribute__((ext_vector_type(16))) __bf16 v16bf;

#define TILE_M 64
#define TILE_N 64
#define TILE_K 32
#define LDP    40   // padded LDS row stride (halfs): 80B = 20 dwords, 16B aligned

// ---------------------------------------------------------------------------
// C = A(f32)[M,K] @ W(f32)[K,N] (+ bias), bf16 WMMA, f32 accum.
// 128 threads = 4 waves; each wave owns a 32x32 sub-tile (2x2 WMMA frags).
// K,N compile-time -> immediate offsets; K-loop fully unrolled with
// register double-buffering so global latency overlaps WMMA work.
// ---------------------------------------------------------------------------
template <int K, int N>
__global__ __launch_bounds__(128)
void gemm_wmma_bf16(const float* __restrict__ A, const float* __restrict__ W,
                    const float* __restrict__ bias, float* __restrict__ C,
                    int M)
{
    __shared__ __bf16 sA[TILE_M][LDP];  // [m][k]
    __shared__ __bf16 sB[TILE_N][LDP];  // transposed: [n][k]

    const int tid  = threadIdx.x;
    const int wave = tid >> 5;
    const int lane = tid & 31;
    const int half = lane >> 4;
    const int l    = lane & 15;
    const int wm   = wave >> 1;
    const int wn   = wave & 1;

    const int bm = blockIdx.y * TILE_M;   // N fastest across blocks: A-tile
    const int bn = blockIdx.x * TILE_N;   // reuse is L2-resident (192MB L2)

    // ---- iteration-invariant staging addresses ----
    // A: thread owns 4 float4 slots (row fixed, k advances by immediates)
    const float* aptr[4];
    float        amask[4];
    int          arow[4], akv[4];
    #pragma unroll
    for (int it = 0; it < 4; ++it) {
        int idx  = tid + it * 128;       // float4 index 0..511
        arow[it] = idx >> 3;             // 0..63
        akv[it]  = (idx & 7) << 2;       // 0,4,...,28
        int grow = bm + arow[it];
        int crow = grow < M ? grow : M - 1;          // clamp, branchless
        aptr[it]  = &A[(size_t)crow * K + akv[it]];
        amask[it] = grow < M ? 1.0f : 0.0f;
    }
    // B: thread owns column n = tid&63, k-half kh = tid>>6 (16 k values)
    const int    bn_n = tid & 63;
    const int    bkh  = tid >> 6;
    const float* wptr = &W[(size_t)(bkh * 16) * N + bn + bn_n];

    // ---- prologue: load chunk 0 into registers (batched clause) ----
    float4 areg[4];
    float  breg[16];
    #pragma unroll
    for (int it = 0; it < 4; ++it) areg[it] = *(const float4*)aptr[it];
    #pragma unroll
    for (int j = 0; j < 16; ++j)   breg[j]  = wptr[j * N];

    v8f acc[2][2] = {};

    #pragma unroll
    for (int k0 = 0; k0 < K; k0 += TILE_K) {
        // ---- regs -> LDS (bf16 convert, packed stores) ----
        #pragma unroll
        for (int it = 0; it < 4; ++it) {
            float4 v = areg[it];
            float  m = amask[it];
            sA[arow[it]][akv[it] + 0] = (__bf16)(v.x * m);
            sA[arow[it]][akv[it] + 1] = (__bf16)(v.y * m);
            sA[arow[it]][akv[it] + 2] = (__bf16)(v.z * m);
            sA[arow[it]][akv[it] + 3] = (__bf16)(v.w * m);
        }
        {
            __bf16 tmp[16];
            #pragma unroll
            for (int j = 0; j < 16; ++j) tmp[j] = (__bf16)breg[j];
            *(v8bf*)&sB[bn_n][bkh * 16 + 0] = *(v8bf*)&tmp[0];
            *(v8bf*)&sB[bn_n][bkh * 16 + 8] = *(v8bf*)&tmp[8];
        }
        __syncthreads();

        // ---- issue next chunk's global loads now (overlap with WMMA) ----
        if (k0 + TILE_K < K) {
            #pragma unroll
            for (int it = 0; it < 4; ++it)
                areg[it] = *(const float4*)(aptr[it] + (k0 + TILE_K));
            #pragma unroll
            for (int j = 0; j < 16; ++j)
                breg[j] = wptr[(size_t)(k0 + TILE_K) * N + j * N];
        }
        if (k0 + 2 * TILE_K < K) {
            #pragma unroll
            for (int it = 0; it < 4; ++it)
                __builtin_prefetch(aptr[it] + (k0 + 2 * TILE_K), 0, 1);
        }

        // ---- fragments: A and B both in contiguous k order, k = 16*half + s ----
        union { v16bf v; v8bf h[2]; } afr[2];
        #pragma unroll
        for (int tm = 0; tm < 2; ++tm) {
            int ar = wm * 32 + tm * 16 + l;
            afr[tm].h[0] = *(const v8bf*)&sA[ar][half * 16];
            afr[tm].h[1] = *(const v8bf*)&sA[ar][half * 16 + 8];
        }
        v16bf bfr[2];
        #pragma unroll
        for (int tn = 0; tn < 2; ++tn) {
            int bc = wn * 32 + tn * 16 + l;
            bfr[tn] = *(const v16bf*)&sB[bc][half * 16];
        }
        #pragma unroll
        for (int tm = 0; tm < 2; ++tm)
            #pragma unroll
            for (int tn = 0; tn < 2; ++tn)
                acc[tm][tn] = __builtin_amdgcn_wmma_f32_16x16x32_bf16(
                    false, afr[tm].v, false, bfr[tn], (short)0, acc[tm][tn],
                    false, false);
        __syncthreads();
    }

    // ---- store C (D layout: vgpr r -> M=r lanes 0-15, M=r+8 lanes 16-31) ----
    #pragma unroll
    for (int tm = 0; tm < 2; ++tm) {
        #pragma unroll
        for (int tn = 0; tn < 2; ++tn) {
            int col = bn + wn * 32 + tn * 16 + l;
            float bv = bias ? bias[col] : 0.0f;
            #pragma unroll
            for (int r = 0; r < 8; ++r) {
                int row = bm + wm * 32 + tm * 16 + half * 8 + r;
                if (row < M)
                    C[(size_t)row * N + col] = acc[tm][tn][r] + bv;
            }
        }
    }
}

// ---------------------------------------------------------------------------
// Degree / norm kernels  (deg starts at 1.0 == self-loop)
// ---------------------------------------------------------------------------
__global__ void deg_init(float* deg, int n) {
    int i = blockIdx.x * blockDim.x + threadIdx.x;
    if (i < n) deg[i] = 1.0f;
}
__global__ void deg_edges(const int* __restrict__ dst, float* deg, int nE) {
    int e = blockIdx.x * blockDim.x + threadIdx.x;
    if (e < nE) atomicAdd(&deg[dst[e]], 1.0f);
}
__global__ void make_dinv(const float* __restrict__ deg, float* dinv, int n) {
    int i = blockIdx.x * blockDim.x + threadIdx.x;
    if (i < n) dinv[i] = rsqrtf(deg[i]);
}

// ---------------------------------------------------------------------------
// agg = hW * dinv^2  (self-loop term; full overwrite)
// ---------------------------------------------------------------------------
__global__ void agg_init(const float* __restrict__ hW,
                         const float* __restrict__ dinv,
                         float* __restrict__ agg, int total) {
    int idx = blockIdx.x * blockDim.x + threadIdx.x;
    if (idx >= total) return;
    float s = dinv[idx >> 8];   // H == 256
    agg[idx] = hW[idx] * s * s;
}

// ---------------------------------------------------------------------------
// Edge scatter: one wave32 per edge. Edge index is wave-uniform ->
// readfirstlane makes src/dst/dinv accesses scalar (constant cache),
// keeping the vector path free for the float4 gathers + f32 atomics.
// ---------------------------------------------------------------------------
__global__ __launch_bounds__(256)
void scatter_edges(const int* __restrict__ src, const int* __restrict__ dst,
                   const float* __restrict__ dinv,
                   const float* __restrict__ hW, float* __restrict__ agg,
                   int nE)
{
    int gt   = blockIdx.x * blockDim.x + threadIdx.x;
    int lane = gt & 31;
    int e    = __builtin_amdgcn_readfirstlane(gt >> 5);   // wave-uniform
    if (e >= nE) return;
    int s = __builtin_amdgcn_readfirstlane(src[e]);
    int d = __builtin_amdgcn_readfirstlane(dst[e]);
    float norm = dinv[s] * dinv[d];
    const float4* hp = (const float4*)&hW[(size_t)s * 256 + lane * 8];
    float4 a = hp[0], b = hp[1];
    float* ap = &agg[(size_t)d * 256 + lane * 8];
    atomicAdd(ap + 0, a.x * norm);
    atomicAdd(ap + 1, a.y * norm);
    atomicAdd(ap + 2, a.z * norm);
    atomicAdd(ap + 3, a.w * norm);
    atomicAdd(ap + 4, b.x * norm);
    atomicAdd(ap + 5, b.y * norm);
    atomicAdd(ap + 6, b.z * norm);
    atomicAdd(ap + 7, b.w * norm);
}

// ---------------------------------------------------------------------------
// out = relu(agg + bias)
// ---------------------------------------------------------------------------
__global__ void finalize_relu(const float* __restrict__ agg,
                              const float* __restrict__ bias,
                              float* __restrict__ outh, int total) {
    int idx = blockIdx.x * blockDim.x + threadIdx.x;
    if (idx >= total) return;
    float v = agg[idx] + bias[idx & 255];
    outh[idx] = v > 0.0f ? v : 0.0f;
}

// ---------------------------------------------------------------------------
// Global mean pool: batch is sorted -> run-length accumulate 64 nodes per
// block (one column per thread), one atomic per (run, column).
// ---------------------------------------------------------------------------
__global__ __launch_bounds__(256)
void pool_accum(const float* __restrict__ h, const int* __restrict__ batch,
                float* __restrict__ pool, int nNodes) {
    int c  = threadIdx.x;              // 0..255 (column)
    int i0 = blockIdx.x * 64;
    int i1 = i0 + 64 < nNodes ? i0 + 64 : nNodes;
    if (i0 >= nNodes) return;
    int   curg = batch[i0];
    float run  = 0.0f;
    for (int i = i0; i < i1; ++i) {
        int g = batch[i];
        if (g != curg) {
            atomicAdd(&pool[curg * 256 + c], run);
            run  = 0.0f;
            curg = g;
        }
        run += h[(size_t)i * 256 + c];
    }
    atomicAdd(&pool[curg * 256 + c], run);
}
__global__ void pool_zero(float* pool, float* cnt) {
    int i = blockIdx.x * blockDim.x + threadIdx.x;
    if (i < 128 * 256) pool[i] = 0.0f;
    if (i < 128) cnt[i] = 0.0f;
}
__global__ void cnt_accum(const int* __restrict__ batch, float* cnt, int n) {
    int i = blockIdx.x * blockDim.x + threadIdx.x;
    if (i < n) atomicAdd(&cnt[batch[i]], 1.0f);
}
__global__ void classifier(const float* __restrict__ pool,
                           const float* __restrict__ cnt,
                           const float* __restrict__ Wout,
                           const float* __restrict__ bout,
                           float* __restrict__ out) {
    int idx = blockIdx.x * blockDim.x + threadIdx.x;  // 0..2047
    if (idx >= 128 * 16) return;
    int g = idx >> 4, j = idx & 15;
    float inv = 1.0f / fmaxf(cnt[g], 1.0f);
    float s = 0.0f;
    for (int k = 0; k < 256; ++k)
        s += pool[g * 256 + k] * Wout[k * 16 + j];
    out[idx] = s * inv + bout[j];
}

// ---------------------------------------------------------------------------
// Driver
// ---------------------------------------------------------------------------
extern "C" void kernel_launch(void* const* d_in, const int* in_sizes, int n_in,
                              void* d_out, int out_size, void* d_ws, size_t ws_size,
                              hipStream_t stream) {
    (void)in_sizes; (void)n_in; (void)out_size; (void)ws_size;

    constexpr int N  = 50000;
    constexpr int E  = 800000;
    constexpr int H  = 256;

    const float* x     = (const float*)d_in[0];
    const int*   ei    = (const int*)d_in[1];     // [2, E] flat
    const int*   batch = (const int*)d_in[2];
    const float* W_in  = (const float*)d_in[3];
    const float* b_in  = (const float*)d_in[4];
    const float* W1    = (const float*)d_in[5];
    const float* b1    = (const float*)d_in[6];
    const float* W2    = (const float*)d_in[7];
    const float* b2    = (const float*)d_in[8];
    const float* Wout  = (const float*)d_in[9];
    const float* bout  = (const float*)d_in[10];
    const int* src = ei;
    const int* dst = ei + E;

    // workspace layout (fp32)
    float* ws   = (float*)d_ws;
    size_t NH   = (size_t)N * H;
    float* B0   = ws;              // h (node features)
    float* B1   = ws + NH;         // hW  (pre-aggregation)
    float* B2   = ws + 2 * NH;     // agg
    float* deg  = ws + 3 * NH;
    float* dinv = deg + N;
    float* pool = dinv + N;
    float* cnt  = pool + 128 * 256;

    const int T = 256;
    dim3 gemm_grid(H / TILE_N, (N + TILE_M - 1) / TILE_M);  // N fastest
    int nodeH_blocks = (int)((NH + T - 1) / T);
    int edge_wave_blocks = (E * 32 + T - 1) / T;

    // degrees + symmetric norm (shared by both GCN layers)
    deg_init<<<(N + T - 1) / T, T, 0, stream>>>(deg, N);
    deg_edges<<<(E + T - 1) / T, T, 0, stream>>>(dst, deg, E);
    make_dinv<<<(N + T - 1) / T, T, 0, stream>>>(deg, dinv, N);

    // h0 = x @ W_in + b_in
    gemm_wmma_bf16<128, 256><<<gemm_grid, 128, 0, stream>>>(x, W_in, b_in, B0, N);

    // ---- GCN layer 1 ----
    gemm_wmma_bf16<256, 256><<<gemm_grid, 128, 0, stream>>>(B0, W1, nullptr, B1, N);
    agg_init<<<nodeH_blocks, T, 0, stream>>>(B1, dinv, B2, (int)NH);
    scatter_edges<<<edge_wave_blocks, T, 0, stream>>>(src, dst, dinv, B1, B2, E);
    finalize_relu<<<nodeH_blocks, T, 0, stream>>>(B2, b1, B0, (int)NH);

    // ---- GCN layer 2 ----
    gemm_wmma_bf16<256, 256><<<gemm_grid, 128, 0, stream>>>(B0, W2, nullptr, B1, N);
    agg_init<<<nodeH_blocks, T, 0, stream>>>(B1, dinv, B2, (int)NH);
    scatter_edges<<<edge_wave_blocks, T, 0, stream>>>(src, dst, dinv, B1, B2, E);
    finalize_relu<<<nodeH_blocks, T, 0, stream>>>(B2, b2, B0, (int)NH);

    // ---- mean pool + classifier ----
    pool_zero<<<(128 * 256 + T - 1) / T, T, 0, stream>>>(pool, cnt);
    pool_accum<<<(N + 63) / 64, T, 0, stream>>>(B0, batch, pool, N);
    cnt_accum<<<(N + T - 1) / T, T, 0, stream>>>(batch, cnt, N);
    classifier<<<(128 * 16 + T - 1) / T, T, 0, stream>>>(pool, cnt, Wout, bout,
                                                         (float*)d_out);
}
